// RecursiveMamba2_PrefixScratchpad_55198919688650
// MI455X (gfx1250) — matrix operand order repaired and checked by reading, not compiled, not run on hardware
//
#include <hip/hip_runtime.h>
#include <hip/hip_bf16.h>
#include <math.h>
#include <stdint.h>

// ---------------- model constants ----------------
#define D_MODEL   2048
#define D_STATE   32
#define HEADDIM   128
#define NHEADS    16
#define D_CONV    4
#define M_PREFIX  8
#define MAX_LOOPS 6
#define BRIDGE_RANK 64
#define VOCAB     50432
#define BSZ       2
#define TLEN      512
#define TE        (M_PREFIX + TLEN)                  // 520
#define ROWS      (BSZ * TE)                         // 1040
#define DIN       (2*D_MODEL + 2*D_STATE + NHEADS)   // 4176
#define CONV_CH   (D_MODEL + 2*D_STATE)              // 2112
#define EPS       1e-6f

// GEMM block tiling (double-buffered 64-wide K stages)
#define BM 64
#define BN 128
#define BK 64
#define ROWS_PAD 1088    // 17 * BM
#define DIN_PAD  4224    // 33 * BN

typedef __bf16 bf16_t;
typedef __attribute__((ext_vector_type(16))) bf16_t v16bf;
typedef __attribute__((ext_vector_type(8)))  bf16_t v8bf;
typedef __attribute__((ext_vector_type(8)))  float  v8f;
typedef __attribute__((ext_vector_type(4)))  unsigned int u32x4;
typedef __attribute__((ext_vector_type(8)))  int i32x8;
typedef __attribute__((ext_vector_type(4)))  int i32x4;

#ifndef __has_builtin
#define __has_builtin(x) 0
#endif
#if defined(__HIP_DEVICE_COMPILE__) && __has_builtin(__builtin_amdgcn_tensor_load_to_lds)
#define HAVE_TDM 1
#else
#define HAVE_TDM 0
#endif

#if HAVE_TDM
// Build a 2D Tensor-DMA descriptor (D# groups 0/1 per CDNA5 ISA §8.3/8.4) and
// issue TENSOR_LOAD_TO_LDS: tile = tileH rows x tileW bf16 elems, row stride
// strideElems, dense row-major into LDS at lds_off.
__device__ __forceinline__ void tdm_load_tile_2d(unsigned lds_off, const bf16_t* gptr,
                                                 int tileW, int tileH, int strideElems)
{
    unsigned long long ga = (unsigned long long)(uintptr_t)gptr;
    u32x4 g0;
    g0[0] = 1u;                                             // count=1 (valid user D#)
    g0[1] = lds_off;                                        // lds_addr
    g0[2] = (unsigned)ga;                                   // global_addr[31:0]
    g0[3] = ((unsigned)(ga >> 32) & 0x01ffffffu)
          | 0x80000000u;                                    // global_addr[56:32] | type=2
    unsigned td0 = (unsigned)tileW, td1 = (unsigned)tileH;
    i32x8 g1;
    g1[0] = (int)(1u << 16);                                // data_size = 1 -> 2 bytes
    g1[1] = (int)((td0 & 0xffffu) << 16);                   // tensor_dim0[15:0]
    g1[2] = (int)(((td0 >> 16) & 0xffffu) | ((td1 & 0xffffu) << 16)); // dim0 hi | dim1 lo
    g1[3] = (int)(((td1 >> 16) & 0xffffu) | ((td0 & 0xffffu) << 16)); // dim1 hi | tile_dim0
    g1[4] = (int)(td1 & 0xffffu);                           // tile_dim1 (tile_dim2=0)
    g1[5] = (int)(unsigned)strideElems;                     // tensor_dim0_stride[31:0]
    g1[6] = 0;                                              // stride hi | dim1_stride lo
    g1[7] = 0;
    i32x4 z4 = {0, 0, 0, 0};
#if __clang_major__ >= 23
    i32x8 z8 = {0, 0, 0, 0, 0, 0, 0, 0};
    __builtin_amdgcn_tensor_load_to_lds(g0, g1, z4, z4, z8, 0);
#else
    __builtin_amdgcn_tensor_load_to_lds(g0, g1, z4, z4, 0);
#endif
}
#endif

// =====================================================================
// bf16 WMMA GEMM: C[M,N] = A[M,K] @ W[N,K]^T (+ optional fp32 addC)
// Block 256 thr = 8 waves -> 64x128 tile; wave = 2x2 wmma tiles (32x32).
// 64-wide K stages double-buffered in LDS: wave 0 issues the Tensor Data
// Mover DMA for stage i+1, then waits tensorcnt<=2 (TDM retires in order,
// so stage i's pair is complete) while other waves run WMMA on stage i.
// Fallback: cooperative b128 copies. bf16 buffers are row-padded so full
// tiles are always in-bounds; stores are element-guarded; tile predicates
// are wave-uniform so EXEC is all-ones at every WMMA.
// Fragment packing per ISA §7.12.2 (16-bit A 16x32, B 32x16, f32 C/D).
// =====================================================================
__global__ __launch_bounds__(256)
void gemm_bf16_wmma_kernel(const bf16_t* __restrict__ A, int lda,
                           const bf16_t* __restrict__ W, int ldw,
                           float* __restrict__ C, int ldc,
                           const float* __restrict__ addC, int ldadd,
                           int M, int N, int K)
{
    __shared__ __align__(16) bf16_t sA[2][BM * BK];   // 2 x 8 KB
    __shared__ __align__(16) bf16_t sB[2][BN * BK];   // 2 x 16 KB
    const int tid  = threadIdx.x;
    const int wid  = tid >> 5;
    const int lane = tid & 31;
    const int wm = wid >> 2, wn = wid & 3;            // 2 x 4 wave grid
    const int r  = lane & 15, kg = lane >> 4;
    const int rowBase = blockIdx.y * BM;
    const int colBase = blockIdx.x * BN;
    const int nstage = K / BK;

    v8f acc[2][2];
    #pragma unroll
    for (int mt = 0; mt < 2; ++mt)
        #pragma unroll
        for (int nt = 0; nt < 2; ++nt) {
            #pragma unroll
            for (int j = 0; j < 8; ++j) acc[mt][nt][j] = 0.f;
            if (addC != nullptr) {
                const int n = colBase + (wn * 2 + nt) * 16 + r;
                #pragma unroll
                for (int j = 0; j < 8; ++j) {
                    const int m = rowBase + (wm * 2 + mt) * 16 + j + (kg << 3);
                    if (m < M && n < N) acc[mt][nt][j] = addC[(size_t)m * ldadd + n];
                }
            }
        }

    // ---- preload stage 0 into buffer 0 ----
#if HAVE_TDM
    if (wid == 0) {
        tdm_load_tile_2d((unsigned)(uintptr_t)&sA[0][0],
                         A + (size_t)rowBase * lda, BK, BM, lda);
        tdm_load_tile_2d((unsigned)(uintptr_t)&sB[0][0],
                         W + (size_t)colBase * ldw, BK, BN, ldw);
    }
#else
    {
        #pragma unroll
        for (int it = 0; it < 2; ++it) {
            const int i = tid + it * 256;
            const int rowA = i >> 3, ca = i & 7;
            int gr = rowBase + rowA; if (gr > M - 1) gr = M - 1;
            *(v8bf*)&sA[0][rowA * BK + ca * 8] =
                *(const v8bf*)&A[(size_t)gr * lda + ca * 8];
        }
        #pragma unroll
        for (int it = 0; it < 4; ++it) {
            const int i = tid + it * 256;
            const int rowB = i >> 3, cb = i & 7;
            int gc = colBase + rowB; if (gc > N - 1) gc = N - 1;
            *(v8bf*)&sB[0][rowB * BK + cb * 8] =
                *(const v8bf*)&W[(size_t)gc * ldw + cb * 8];
        }
    }
#endif

    for (int st = 0; st < nstage; ++st) {
        const int cur = st & 1, nxt = cur ^ 1;
        const int k1 = (st + 1) * BK;
        __syncthreads();   // (a) everyone done computing from buf[nxt]
#if HAVE_TDM
        if (wid == 0) {
            if (st + 1 < nstage) {
                tdm_load_tile_2d((unsigned)(uintptr_t)&sA[nxt][0],
                                 A + (size_t)rowBase * lda + k1, BK, BM, lda);
                tdm_load_tile_2d((unsigned)(uintptr_t)&sB[nxt][0],
                                 W + (size_t)colBase * ldw + k1, BK, BN, ldw);
                __builtin_amdgcn_s_wait_tensorcnt(2);   // stage st complete (in-order)
            } else {
                __builtin_amdgcn_s_wait_tensorcnt(0);
            }
        }
#else
        if (st + 1 < nstage) {
            #pragma unroll
            for (int it = 0; it < 2; ++it) {
                const int i = tid + it * 256;
                const int rowA = i >> 3, ca = i & 7;
                int gr = rowBase + rowA; if (gr > M - 1) gr = M - 1;
                *(v8bf*)&sA[nxt][rowA * BK + ca * 8] =
                    *(const v8bf*)&A[(size_t)gr * lda + k1 + ca * 8];
            }
            #pragma unroll
            for (int it = 0; it < 4; ++it) {
                const int i = tid + it * 256;
                const int rowB = i >> 3, cb = i & 7;
                int gc = colBase + rowB; if (gc > N - 1) gc = N - 1;
                *(v8bf*)&sB[nxt][rowB * BK + cb * 8] =
                    *(const v8bf*)&W[(size_t)gc * ldw + k1 + cb * 8];
            }
        }
#endif
        __syncthreads();   // (b) buf[cur] published to all waves

        const bf16_t* curA = &sA[cur][0];
        const bf16_t* curB = &sB[cur][0];
        #pragma unroll
        for (int s = 0; s < 2; ++s) {                 // two 32-K substeps
            v16bf af[2], bfr[2];
            #pragma unroll
            for (int mt = 0; mt < 2; ++mt) {
                const int arow = (wm * 2 + mt) * 16 + r;
                v8bf lo = *(const v8bf*)&curA[arow * BK + 32 * s + (kg << 3)];
                v8bf hi = *(const v8bf*)&curA[arow * BK + 32 * s + 16 + (kg << 3)];
                af[mt] = __builtin_shufflevector(lo, hi,
                         0,1,2,3,4,5,6,7,8,9,10,11,12,13,14,15);
            }
            #pragma unroll
            for (int nt = 0; nt < 2; ++nt) {
                const int brow = (wn * 2 + nt) * 16 + r;
                v8bf lo = *(const v8bf*)&curB[brow * BK + 32 * s + (kg << 3)];
                v8bf hi = *(const v8bf*)&curB[brow * BK + 32 * s + 16 + (kg << 3)];
                bfr[nt] = __builtin_shufflevector(lo, hi,
                          0,1,2,3,4,5,6,7,8,9,10,11,12,13,14,15);
            }
            #pragma unroll
            for (int mt = 0; mt < 2; ++mt)
                #pragma unroll
                for (int nt = 0; nt < 2; ++nt)
                    acc[mt][nt] = __builtin_amdgcn_wmma_f32_16x16x32_bf16(
                        false, af[mt], false, bfr[nt], (short)0, acc[mt][nt], false, false);
        }
    }

    #pragma unroll
    for (int mt = 0; mt < 2; ++mt)
        #pragma unroll
        for (int nt = 0; nt < 2; ++nt) {
            const int n = colBase + (wn * 2 + nt) * 16 + r;
            #pragma unroll
            for (int j = 0; j < 8; ++j) {
                const int m = rowBase + (wm * 2 + mt) * 16 + j + (kg << 3);
                if (m < M && n < N) C[(size_t)m * ldc + n] = acc[mt][nt][j];
            }
        }
}

// ---------------- fp32 -> bf16 (+ zero row padding) ----------------
__global__ void cvt_pad_bf16_kernel(const float* __restrict__ src, bf16_t* __restrict__ dst,
                                    int srcRows, int dstRows, int width)
{
    int idx = blockIdx.x * blockDim.x + threadIdx.x;
    if (idx >= dstRows * width) return;
    int row = idx / width, c = idx % width;
    dst[idx] = (row < srcRows) ? (bf16_t)src[(size_t)row * width + c] : (bf16_t)0.f;
}

__global__ void zero_bf16_kernel(bf16_t* __restrict__ p, int n)
{
    int idx = blockIdx.x * blockDim.x + threadIdx.x;
    if (idx < n) p[idx] = (bf16_t)0.f;
}

// -------- LoRA fuse to bf16: W = bf16(base + 2*(B @ A)), zero-padded rows --------
__global__ void lora_bf16_kernel(const float* __restrict__ base,
                                 const float* __restrict__ Bm,   // (rows, 4)
                                 const float* __restrict__ Am,   // (4, D_MODEL)
                                 bf16_t* __restrict__ Wo, int rows, int dstRows)
{
    int idx = blockIdx.x * blockDim.x + threadIdx.x;
    if (idx >= dstRows * D_MODEL) return;
    int i = idx / D_MODEL, j = idx % D_MODEL;
    float s = 0.f;
    if (i < rows) {
        s = base[(size_t)i * D_MODEL + j];
        #pragma unroll
        for (int rr = 0; rr < 4; ++rr)
            s += 2.0f * Bm[i * 4 + rr] * Am[rr * D_MODEL + j];
    }
    Wo[idx] = (bf16_t)s;
}

// ---------------- initial xe = concat(latent_memory, emb[ids]) ----------------
__global__ void init_xe_kernel(const int* __restrict__ ids,
                               const float* __restrict__ emb,
                               const float* __restrict__ latent,
                               float* __restrict__ xe)
{
    int idx = blockIdx.x * blockDim.x + threadIdx.x;
    if (idx >= ROWS * D_MODEL) return;
    int row = idx / D_MODEL, c = idx % D_MODEL;
    int b = row / TE, t = row % TE;
    if (t < M_PREFIX) xe[idx] = latent[t * D_MODEL + c];
    else              xe[idx] = emb[(size_t)ids[b * TLEN + (t - M_PREFIX)] * D_MODEL + c];
}

// -------- per-loop: xe[:,M:] += gate * emb[ids]; rope(xe, loop_i); bf16 copy --------
__global__ void lifeline_rope_kernel(float* __restrict__ xe,
                                     bf16_t* __restrict__ xe_bf,     // ROWS_PAD rows
                                     const float* __restrict__ emb,
                                     const int* __restrict__ ids,
                                     const float* __restrict__ gate,
                                     int loop_i)
{
    int idx = blockIdx.x * blockDim.x + threadIdx.x;   // ROWS_PAD * 1024 pairs
    if (idx >= ROWS_PAD * (D_MODEL / 2)) return;
    int row = idx >> 10, pair = idx & 1023;
    int c0 = 2 * pair, c1 = c0 + 1;
    size_t base = (size_t)row * D_MODEL;
    if (row >= ROWS) {                                  // padded tail rows -> zero bf16
        xe_bf[base + c0] = (bf16_t)0.f;
        xe_bf[base + c1] = (bf16_t)0.f;
        return;
    }
    int b = row / TE, t = row % TE;
    float x1 = xe[base + c0], x2 = xe[base + c1];
    if (t >= M_PREFIX) {
        size_t e = (size_t)ids[b * TLEN + (t - M_PREFIX)] * D_MODEL;
        x1 += gate[c0] * emb[e + c0];
        x2 += gate[c1] * emb[e + c1];
    }
    float f = (float)loop_i * __expf(-logf(10000.f) * ((float)pair / 1024.f));
    float cs = cosf(f), sn = sinf(f);
    float o0 = x1 * cs - x2 * sn;
    float o1 = x2 * cs + x1 * sn;
    xe[base + c0] = o0;            xe[base + c1] = o1;
    xe_bf[base + c0] = (bf16_t)o0; xe_bf[base + c1] = (bf16_t)o1;
}

// -------- causal depthwise conv (K=4) over time + bias + silu --------
__global__ void conv_silu_kernel(const float* __restrict__ zx,
                                 const float* __restrict__ conv_w,   // (2112,4)
                                 const float* __restrict__ conv_b,
                                 float* __restrict__ xconv)
{
    int idx = blockIdx.x * blockDim.x + threadIdx.x;
    if (idx >= ROWS * CONV_CH) return;
    int row = idx / CONV_CH, ch = idx % CONV_CH;
    int b = row / TE, t = row % TE;
    float acc = 0.f;
    #pragma unroll
    for (int k = 0; k < D_CONV; ++k) {
        int ts = t - (D_CONV - 1) + k;
        if (ts >= 0)
            acc += zx[(size_t)(b * TE + ts) * DIN + D_MODEL + ch] * conv_w[ch * D_CONV + k];
    }
    acc += conv_b[ch];
    xconv[idx] = acc / (1.f + __expf(-acc));
}

// -------- dt = softplus(raw + bias); dA = exp(-exp(A_log)*dt) --------
__global__ void dt_kernel(const float* __restrict__ zx,
                          const float* __restrict__ dt_bias,
                          const float* __restrict__ A_log,
                          float* __restrict__ dtb, float* __restrict__ dAb)
{
    int idx = blockIdx.x * blockDim.x + threadIdx.x;
    if (idx >= ROWS * NHEADS) return;
    int row = idx / NHEADS, h = idx % NHEADS;
    float raw = zx[(size_t)row * DIN + (2 * D_MODEL + 2 * D_STATE) + h] + dt_bias[h];
    float dtv = (raw > 20.f) ? raw : log1pf(__expf(raw));
    dtb[idx] = dtv;
    dAb[idx] = __expf(-__expf(A_log[h]) * dtv);
}

// -------- sequential SSM scan: block = (b,h); thread = p; 32-wide state in regs --------
__global__ __launch_bounds__(HEADDIM)
void ssm_scan_kernel(const float* __restrict__ xconv,
                     const float* __restrict__ dAb,
                     const float* __restrict__ dtb,
                     const float* __restrict__ D_skip,
                     float* __restrict__ y)
{
    __shared__ float sB[D_STATE];
    __shared__ float sC[D_STATE];
    __shared__ float sS[2];
    const int bh = blockIdx.x;
    const int b = bh / NHEADS, h = bh % NHEADS;
    const int p = threadIdx.x;
    float hreg[D_STATE];
    #pragma unroll
    for (int n = 0; n < D_STATE; ++n) hreg[n] = 0.f;
    const float dsk = D_skip[h];

    for (int t = 0; t < TE; ++t) {
        const int row = b * TE + t;
        const float* xrow = xconv + (size_t)row * CONV_CH;
        if (p < D_STATE)          sB[p]           = xrow[D_MODEL + p];
        else if (p < 2 * D_STATE) sC[p - D_STATE] = xrow[D_MODEL + p];
        else if (p == 64)         sS[0]           = dAb[row * NHEADS + h];
        else if (p == 65)         sS[1]           = dtb[row * NHEADS + h];
        __syncthreads();
        const float x = xrow[h * HEADDIM + p];
        const float a = sS[0];
        const float coef = sS[1] * x;
        float acc = 0.f;
        #pragma unroll
        for (int n = 0; n < D_STATE; ++n) {
            hreg[n] = a * hreg[n] + coef * sB[n];
            acc += hreg[n] * sC[n];
        }
        y[(size_t)row * D_MODEL + h * HEADDIM + p] = acc + dsk * x;
        __syncthreads();
    }
}

// -------- ybf = bf16(rmsnorm(y * silu(z), ssm_norm_w)) --------
__global__ __launch_bounds__(256)
void gate_rmsnorm_kernel(const float* __restrict__ y,
                         const float* __restrict__ zx,
                         const float* __restrict__ w,
                         bf16_t* __restrict__ out_bf)
{
    const int row = blockIdx.x, tid = threadIdx.x;
    const float* zrow = zx + (size_t)row * DIN;
    const float* yrow = y + (size_t)row * D_MODEL;
    float v[8]; float ss = 0.f;
    #pragma unroll
    for (int i = 0; i < 8; ++i) {
        int c = tid + 256 * i;
        float z = zrow[c];
        float g = z / (1.f + __expf(-z));
        float t = yrow[c] * g;
        v[i] = t; ss += t * t;
    }
    __shared__ float red[256];
    red[tid] = ss; __syncthreads();
    for (int s = 128; s > 0; s >>= 1) { if (tid < s) red[tid] += red[tid + s]; __syncthreads(); }
    float inv = rsqrtf(red[0] / (float)D_MODEL + EPS);
    #pragma unroll
    for (int i = 0; i < 8; ++i) {
        int c = tid + 256 * i;
        out_bf[(size_t)row * D_MODEL + c] = (bf16_t)(v[i] * inv * w[c]);
    }
}

// -------- out = rmsnorm(in, w) -> fp32 + bf16 --------
__global__ __launch_bounds__(256)
void rmsnorm_kernel(const float* __restrict__ in,
                    const float* __restrict__ w,
                    float* __restrict__ outf,
                    bf16_t* __restrict__ outbf)
{
    const int row = blockIdx.x, tid = threadIdx.x;
    const float* irow = in + (size_t)row * D_MODEL;
    float v[8]; float ss = 0.f;
    #pragma unroll
    for (int i = 0; i < 8; ++i) { int c = tid + 256 * i; float t = irow[c]; v[i] = t; ss += t * t; }
    __shared__ float red[256];
    red[tid] = ss; __syncthreads();
    for (int s = 128; s > 0; s >>= 1) { if (tid < s) red[tid] += red[tid + s]; __syncthreads(); }
    float inv = rsqrtf(red[0] / (float)D_MODEL + EPS);
    #pragma unroll
    for (int i = 0; i < 8; ++i) {
        int c = tid + 256 * i;
        float o = v[i] * inv * w[c];
        outf[(size_t)row * D_MODEL + c] = o;
        outbf[(size_t)row * D_MODEL + c] = (bf16_t)o;
    }
}

// ---------------- host-side GEMM launcher ----------------
static void launch_gemm(const bf16_t* A, int lda, const bf16_t* W, int ldw,
                        float* C, int ldc, const float* addC, int ldadd,
                        int M, int N, int K, hipStream_t s)
{
    dim3 grid((N + BN - 1) / BN, (M + BM - 1) / BM);
    gemm_bf16_wmma_kernel<<<grid, 256, 0, s>>>(A, lda, W, ldw, C, ldc, addC, ldadd, M, N, K);
}

extern "C" void kernel_launch(void* const* d_in, const int* in_sizes, int n_in,
                              void* d_out, int out_size, void* d_ws, size_t ws_size,
                              hipStream_t stream)
{
    (void)in_sizes; (void)n_in; (void)out_size; (void)ws_size;
    const int*   ids         = (const int*)  d_in[0];
    const float* emb         = (const float*)d_in[1];
    const float* latent      = (const float*)d_in[2];
    const float* gate        = (const float*)d_in[3];
    const float* loop_norm_w = (const float*)d_in[4];
    const float* in_base     = (const float*)d_in[5];
    const float* in_A        = (const float*)d_in[6];
    const float* in_B        = (const float*)d_in[7];
    const float* conv_w      = (const float*)d_in[8];
    const float* conv_b      = (const float*)d_in[9];
    const float* dt_bias     = (const float*)d_in[10];
    const float* A_log       = (const float*)d_in[11];
    const float* D_skip      = (const float*)d_in[12];
    const float* ssm_norm_w  = (const float*)d_in[13];
    const float* out_base    = (const float*)d_in[14];
    const float* out_A       = (const float*)d_in[15];
    const float* out_B       = (const float*)d_in[16];
    const float* bridge_down = (const float*)d_in[17];
    const float* bridge_up   = (const float*)d_in[18];
    float* out = (float*)d_out;

    // ---- workspace carve-out (256B-aligned chunks) ----
    char* wsb = (char*)d_ws;
    size_t off = 0;
    auto carve = [&](size_t bytes) -> char* {
        char* p = wsb + off;
        off += (bytes + 255) & ~(size_t)255;
        return p;
    };
    bf16_t* emb_bf  = (bf16_t*)carve((size_t)VOCAB * D_MODEL * 2);
    bf16_t* Win_bf  = (bf16_t*)carve((size_t)DIN_PAD * D_MODEL * 2);
    bf16_t* Wout_bf = (bf16_t*)carve((size_t)D_MODEL * D_MODEL * 2);
    bf16_t* bdn_bf  = (bf16_t*)carve((size_t)BN * D_MODEL * 2);          // 64 -> 128 rows
    bf16_t* bup_bf  = (bf16_t*)carve((size_t)D_MODEL * BRIDGE_RANK * 2);
    float*  xe      = (float*) carve((size_t)ROWS * D_MODEL * 4);
    bf16_t* xe_bf   = (bf16_t*)carve((size_t)ROWS_PAD * D_MODEL * 2);
    float*  zx      = (float*) carve((size_t)ROWS * DIN * 4);
    float*  xconv   = (float*) carve((size_t)ROWS * CONV_CH * 4);
    float*  dtb     = (float*) carve((size_t)ROWS * NHEADS * 4);
    float*  dAb     = (float*) carve((size_t)ROWS * NHEADS * 4);
    float*  ybuf    = (float*) carve((size_t)ROWS * D_MODEL * 4);
    bf16_t* ybuf_bf = (bf16_t*)carve((size_t)ROWS_PAD * D_MODEL * 2);
    float*  tmp     = (float*) carve((size_t)ROWS * D_MODEL * 4);
    float*  bmid    = (float*) carve((size_t)ROWS * BRIDGE_RANK * 4);
    bf16_t* bmid_bf = (bf16_t*)carve((size_t)ROWS_PAD * BRIDGE_RANK * 2);
    float*  xb      = (float*) carve((size_t)ROWS * D_MODEL * 4);
    bf16_t* xb_bf   = (bf16_t*)carve((size_t)ROWS * D_MODEL * 2);

    // ---- one-time conversions / weight fusion ----
    cvt_pad_bf16_kernel<<<((size_t)VOCAB * D_MODEL + 255) / 256, 256, 0, stream>>>(
        emb, emb_bf, VOCAB, VOCAB, D_MODEL);
    lora_bf16_kernel<<<((size_t)DIN_PAD * D_MODEL + 255) / 256, 256, 0, stream>>>(
        in_base, in_B, in_A, Win_bf, DIN, DIN_PAD);
    lora_bf16_kernel<<<((size_t)D_MODEL * D_MODEL + 255) / 256, 256, 0, stream>>>(
        out_base, out_B, out_A, Wout_bf, D_MODEL, D_MODEL);
    cvt_pad_bf16_kernel<<<((size_t)BN * D_MODEL + 255) / 256, 256, 0, stream>>>(
        bridge_down, bdn_bf, BRIDGE_RANK, BN, D_MODEL);
    cvt_pad_bf16_kernel<<<((size_t)D_MODEL * BRIDGE_RANK + 255) / 256, 256, 0, stream>>>(
        bridge_up, bup_bf, D_MODEL, D_MODEL, BRIDGE_RANK);
    zero_bf16_kernel<<<((ROWS_PAD - ROWS) * D_MODEL + 255) / 256, 256, 0, stream>>>(
        ybuf_bf + (size_t)ROWS * D_MODEL, (ROWS_PAD - ROWS) * D_MODEL);

    // ---- xe0 = [latent ; emb[ids]] ----
    init_xe_kernel<<<(ROWS * D_MODEL + 255) / 256, 256, 0, stream>>>(ids, emb, latent, xe);

    for (int loop = 0; loop < MAX_LOOPS; ++loop) {
        lifeline_rope_kernel<<<(ROWS_PAD * (D_MODEL / 2) + 255) / 256, 256, 0, stream>>>(
            xe, xe_bf, emb, ids, gate, loop);
        // zxbcdt = xe @ Win^T
        launch_gemm(xe_bf, D_MODEL, Win_bf, D_MODEL, zx, DIN, nullptr, 0,
                    ROWS, DIN, D_MODEL, stream);
        conv_silu_kernel<<<(ROWS * CONV_CH + 255) / 256, 256, 0, stream>>>(zx, conv_w, conv_b, xconv);
        dt_kernel<<<(ROWS * NHEADS + 255) / 256, 256, 0, stream>>>(zx, dt_bias, A_log, dtb, dAb);
        ssm_scan_kernel<<<BSZ * NHEADS, HEADDIM, 0, stream>>>(xconv, dAb, dtb, D_skip, ybuf);
        gate_rmsnorm_kernel<<<ROWS, 256, 0, stream>>>(ybuf, zx, ssm_norm_w, ybuf_bf);
        // tmp = xe + y @ Wout^T
        launch_gemm(ybuf_bf, D_MODEL, Wout_bf, D_MODEL, tmp, D_MODEL, xe, D_MODEL,
                    ROWS, D_MODEL, D_MODEL, stream);
        rmsnorm_kernel<<<ROWS, 256, 0, stream>>>(tmp, loop_norm_w, xe, xe_bf);
    }

    // ---- bridge: xb = xe + (xe @ down^T) @ up^T ----
    launch_gemm(xe_bf, D_MODEL, bdn_bf, D_MODEL, bmid, BRIDGE_RANK, nullptr, 0,
                ROWS, BRIDGE_RANK, D_MODEL, stream);
    cvt_pad_bf16_kernel<<<((size_t)ROWS_PAD * BRIDGE_RANK + 255) / 256, 256, 0, stream>>>(
        bmid, bmid_bf, ROWS, ROWS_PAD, BRIDGE_RANK);
    launch_gemm(bmid_bf, BRIDGE_RANK, bup_bf, BRIDGE_RANK, xb, D_MODEL, xe, D_MODEL,
                ROWS, D_MODEL, BRIDGE_RANK, stream);
    cvt_pad_bf16_kernel<<<((size_t)ROWS * D_MODEL + 255) / 256, 256, 0, stream>>>(
        xb, xb_bf, ROWS, ROWS, D_MODEL);

    // ---- logits = xb[:, M:] @ emb^T (per batch: 512 rows, exact 8x394 tile grid) ----
    for (int b = 0; b < BSZ; ++b) {
        launch_gemm(xb_bf + (size_t)(b * TE + M_PREFIX) * D_MODEL, D_MODEL,
                    emb_bf, D_MODEL,
                    out + (size_t)b * TLEN * VOCAB, VOCAB, nullptr, 0,
                    TLEN, VOCAB, D_MODEL, stream);
    }
}
